// Layer3_62551903699074
// MI455X (gfx1250) — matrix operand order repaired and checked by reading, not compiled, not run on hardware
//
#include <hip/hip_runtime.h>

typedef __attribute__((ext_vector_type(16))) __bf16 v16bf;
typedef __attribute__((ext_vector_type(8)))  float  v8f;

#define LAMB_CONST 0.2f

__device__ __forceinline__ __bf16 f2bf(float f) {
    unsigned u = __float_as_uint(f);
    u += 0x7fffu + ((u >> 16) & 1u);           // round-to-nearest-even
    unsigned short h = (unsigned short)(u >> 16);
    return __builtin_bit_cast(__bf16, h);
}

// ---------------- pre-pass: pack weights  wp[(tap*768+co)*768+ci] = bf16(w[(ci*768+co)*9+tap])
__global__ __launch_bounds__(256) void pack_weights(const float* __restrict__ w,
                                                    __bf16* __restrict__ wp) {
    size_t o = (size_t)blockIdx.x * 256 + threadIdx.x;     // 5,308,416 total
    int ci  = (int)(o % 768);
    size_t t = o / 768;
    int co  = (int)(t % 768);
    int tap = (int)(t / 768);
    wp[o] = f2bf(w[((size_t)ci * 768 + co) * 9 + tap]);
}

// ---------------- pre-pass: f32 NCHW -> bf16 with zero halo (PD = IMG+2)
template<int IMG>
__global__ __launch_bounds__(256) void pad_input(const float* __restrict__ in,
                                                 __bf16* __restrict__ out) {
    constexpr int PD = IMG + 2;
    size_t o = (size_t)blockIdx.x * 256 + threadIdx.x;
    int ix  = (int)(o % PD);
    size_t t = o / PD;
    int iy  = (int)(t % PD);
    size_t bc = t / PD;                                    // fused batch*channel
    float v = 0.f;
    if (iy >= 1 && iy <= IMG && ix >= 1 && ix <= IMG)
        v = in[(bc * IMG + (iy - 1)) * IMG + (ix - 1)];
    out[o] = f2bf(v);
}

// O[b,co,y,x] = sum_{ci,ky,kx} W[ci,co,ky,kx] * I[b,ci, y+1-ky, x+1-kx]
// EPI: 0 -> out0 = conv ; 1 -> out0 = out1 = LAMB*exp(-conv)*|aux| ; 2 -> out0 = |aux - conv|
template<int IMG, int NPX, int NT, int EPI>
__global__ __launch_bounds__(256)
void conv3x3_convT_wmma(const __bf16* __restrict__ inp,  // padded bf16 (32*768, PD, PD)
                        const __bf16* __restrict__ wp,   // packed bf16 (9, 768, 768)
                        const float*  __restrict__ aux,
                        float* __restrict__ out0,
                        float* __restrict__ out1)
{
    constexpr int CI = 768, CO = 768;
    constexpr int CO_WG = 128;
    constexpr int KC = 32;
    constexpr int NCHUNK = CI / KC;            // 24
    constexpr int TOT = 9 * NCHUNK;            // 216 K-steps
    constexpr int N_SUB = NT * 16;
    constexpr int APAD = 40;                   // 80 B row stride, 16B-aligned
    constexpr int PD = IMG + 2;

    __shared__ __align__(16) __bf16 As[2][CO_WG][APAD];
    __shared__ __align__(16) __bf16 Bs[2][NPX][APAD];

    const int tid   = threadIdx.x;
    const int lane  = tid & 31;
    const int wave  = tid >> 5;
    const int wm    = wave >> 1;
    const int wn    = wave & 1;
    const int l15   = lane & 15;
    const int lhalf = lane >> 4;

    constexpr int nPerImg = (IMG * IMG) / NPX;
    const int m   = blockIdx.x / (32 * nPerImg);
    const int rem = blockIdx.x % (32 * nPerImg);
    const int b   = rem / nPerImg;
    const int nb  = rem % nPerImg;
    const int co0 = m * CO_WG;
    const int px0 = nb * NPX;
    const size_t bci0 = (size_t)b * CI;

    // stage K-step s into LDS buffer `buf`
    auto stage = [&](int s, int buf) {
        const int tap = s / NCHUNK;
        const int kc  = s % NCHUNK;
        // ---- A: 8KB tile of packed weights, direct global->LDS async b128
        const __bf16* gA = wp + ((size_t)tap * CO + co0) * CI + kc * KC;
        #pragma unroll
        for (int i = 0; i < 2; ++i) {
            int idx = tid + i * 256;           // 512 x uint4
            int row = idx >> 2;
            int c8  = idx & 3;
            const void* g = gA + (size_t)row * CI + c8 * 8;
            unsigned l = (unsigned)(uintptr_t)&As[buf][row][c8 * 8];  // LDS addr = low 32 bits
            asm volatile("global_load_async_to_lds_b128 %0, %1, off"
                         :: "v"(l), "v"(g) : "memory");
        }
        // ---- B: halo-free gather from padded bf16 image (2B loads)
        const int oy = 2 - tap / 3;            // padded row offset for this tap
        const int ox = 2 - tap % 3;
        #pragma unroll
        for (int i = 0; i < (NPX * KC) / 256; ++i) {
            int idx = tid + i * 256;
            int pxl = idx % NPX;
            int kr  = idx / NPX;
            int ci  = kc * KC + kr;
            int p   = px0 + pxl;
            int py  = p / IMG, px = p % IMG;
            Bs[buf][pxl][kr] = inp[((bci0 + ci) * PD + py + oy) * PD + px + ox];
        }
    };

    v8f acc[2][NT];
    #pragma unroll
    for (int mt = 0; mt < 2; ++mt)
        #pragma unroll
        for (int nt = 0; nt < NT; ++nt)
            acc[mt][nt] = (v8f){0.f, 0.f, 0.f, 0.f, 0.f, 0.f, 0.f, 0.f};

    stage(0, 0);
    asm volatile("s_wait_asynccnt 0x0" ::: "memory");
    __syncthreads();

    for (int s = 0; s < TOT; ++s) {
        const int cur = s & 1;
        if (s + 1 < TOT) stage(s + 1, cur ^ 1);   // overlap next-chunk staging with WMMA

        v16bf afrag[2];
        #pragma unroll
        for (int mt = 0; mt < 2; ++mt) {
            const uint4* ap = (const uint4*)(&As[cur][wm * 32 + mt * 16 + l15][0]);
            union { v16bf v; uint4 q[2]; } au;
            au.q[0] = ap[lhalf];                  // K = 8*lhalf .. +7
            au.q[1] = ap[2 + lhalf];              // K = 16+8*lhalf .. +7
            afrag[mt] = au.v;
        }
        v16bf bfrag[NT];
        #pragma unroll
        for (int nt = 0; nt < NT; ++nt) {
            const uint4* bp = (const uint4*)(&Bs[cur][wn * N_SUB + nt * 16 + l15][0]);
            union { v16bf v; uint4 q[2]; } bu;
            bu.q[0] = bp[lhalf];                  // K = 8*lhalf .. +7
            bu.q[1] = bp[2 + lhalf];              // K = 16+8*lhalf .. +7
            bfrag[nt] = bu.v;
        }
        #pragma unroll
        for (int mt = 0; mt < 2; ++mt)
            #pragma unroll
            for (int nt = 0; nt < NT; ++nt)
                acc[mt][nt] = __builtin_amdgcn_wmma_f32_16x16x32_bf16(
                    false, afrag[mt], false, bfrag[nt],
                    (short)0, acc[mt][nt], false, false);

        asm volatile("s_wait_asynccnt 0x0" ::: "memory");   // next A tile landed
        __syncthreads();                                    // + next B tile visible
    }

    // ---- fused epilogue; D layout: lane(l15,lhalf): N=l15, M=r+8*lhalf in vgpr r
    #pragma unroll
    for (int mt = 0; mt < 2; ++mt) {
        #pragma unroll
        for (int nt = 0; nt < NT; ++nt) {
            const int p = px0 + wn * N_SUB + nt * 16 + l15;
            #pragma unroll
            for (int r = 0; r < 8; ++r) {
                int co = co0 + wm * 32 + mt * 16 + r + 8 * lhalf;
                size_t oidx = (((size_t)b * CO + co) * IMG + p / IMG) * IMG + (p % IMG);
                float v = acc[mt][nt][r];
                if constexpr (EPI == 0) {
                    out0[oidx] = v;
                } else if constexpr (EPI == 1) {
                    float rv = LAMB_CONST * __expf(-v) * fabsf(aux[oidx]);
                    out0[oidx] = rv;
                    out1[oidx] = rv;
                } else {
                    out0[oidx] = fabsf(aux[oidx] - v);
                }
            }
        }
    }
}

extern "C" void kernel_launch(void* const* d_in, const int* in_sizes, int n_in,
                              void* d_out, int out_size, void* d_ws, size_t ws_size,
                              hipStream_t stream) {
    (void)in_sizes; (void)n_in; (void)out_size; (void)ws_size;
    // setup_inputs order: img_batch, x, u, xp, Rt_1, U_w, B_w, A_w
    const float* x   = (const float*)d_in[1];
    const float* u   = (const float*)d_in[2];
    const float* xp  = (const float*)d_in[3];
    const float* Rt1 = (const float*)d_in[4];
    const float* Uw  = (const float*)d_in[5];
    const float* Bw  = (const float*)d_in[6];
    const float* Aw  = (const float*)d_in[7];

    float* out = (float*)d_out;
    const size_t big   = (size_t)32 * 768 * 16 * 16;  // 6,291,456
    const size_t small = (size_t)32 * 768 * 8 * 8;    // 1,572,864
    float* pred  = out;
    float* out2  = out + big;
    float* predx = out + big + small;
    float* out4  = out + big + small + big;

    // workspace layout (bf16): 3 packed weights + 2 padded 16x16 inputs + 1 padded 8x8 input
    const size_t WN   = (size_t)9 * 768 * 768;         // 5,308,416 per weight
    const size_t PAD16 = (size_t)32 * 768 * 18 * 18;   // 7,962,624
    const size_t PAD8  = (size_t)32 * 768 * 10 * 10;   // 2,457,600
    __bf16* wsb  = (__bf16*)d_ws;
    __bf16* UwP  = wsb;
    __bf16* BwP  = UwP + WN;
    __bf16* AwP  = BwP + WN;
    __bf16* xP   = AwP + WN;
    __bf16* RtP  = xP + PAD16;
    __bf16* uP   = RtP + PAD16;                        // total ~65.5 MB

    pack_weights<<<dim3((unsigned)(WN / 256)), dim3(256), 0, stream>>>(Uw, UwP);
    pack_weights<<<dim3((unsigned)(WN / 256)), dim3(256), 0, stream>>>(Bw, BwP);
    pack_weights<<<dim3((unsigned)(WN / 256)), dim3(256), 0, stream>>>(Aw, AwP);
    pad_input<16><<<dim3((unsigned)(PAD16 / 256)), dim3(256), 0, stream>>>(x,   xP);
    pad_input<16><<<dim3((unsigned)(PAD16 / 256)), dim3(256), 0, stream>>>(Rt1, RtP);
    pad_input<8> <<<dim3((unsigned)(PAD8  / 256)), dim3(256), 0, stream>>>(u,   uP);

    // pred = convT(x, U_w)
    conv3x3_convT_wmma<16, 128, 4, 0>
        <<<dim3(6 * 32 * 2), dim3(256), 0, stream>>>(xP, UwP, nullptr, pred, nullptr);
    // out2 = out4 = LAMB * exp(-convT(u, B_w)) * |xp|
    conv3x3_convT_wmma<8, 64, 2, 1>
        <<<dim3(6 * 32), dim3(256), 0, stream>>>(uP, BwP, xp, out2, out4);
    // pred_xt_1 = |x - convT(Rt_1, A_w)|
    conv3x3_convT_wmma<16, 128, 4, 2>
        <<<dim3(6 * 32 * 2), dim3(256), 0, stream>>>(RtP, AwP, x, predx, nullptr);
}
